// HealthAttention_12481174962384
// MI455X (gfx1250) — matrix operand order repaired and checked by reading, not compiled, not run on hardware
//
#include <hip/hip_runtime.h>

// ---------------------------------------------------------------------------
// Problem constants (B=2, S=2048, D=2048, H=16, DH=128)
// ---------------------------------------------------------------------------
#define CB 2
#define CS 2048
#define CD 2048
#define CH 16
#define CDH 128
#define CM (CB * CS) // 4096 rows total

typedef __bf16 bf16_t;
typedef unsigned short u16;
typedef bf16_t v16bf __attribute__((ext_vector_type(16)));
typedef bf16_t v8bf  __attribute__((ext_vector_type(8)));
typedef float  v8f   __attribute__((ext_vector_type(8)));

// fp32 -> bf16 (RNE) raw bits
__device__ __forceinline__ u16 f2bfu(float f) {
    unsigned u = __builtin_bit_cast(unsigned, f);
    u += 0x7fffu + ((u >> 16) & 1u);
    return (u16)(u >> 16);
}
__device__ __forceinline__ bf16_t f2bf(float f) {
    u16 s = f2bfu(f);
    return __builtin_bit_cast(bf16_t, s);
}
__device__ __forceinline__ float bf2f(bf16_t b) {
    unsigned u = (unsigned)__builtin_bit_cast(u16, b) << 16;
    return __builtin_bit_cast(float, u);
}

// A-operand loader (16-bit A 16x32 layout): lane holds row = lane%16.
// Element e<8 -> K = 8*half + e ; e>=8 -> K = 16 + 8*half + (e-8).
__device__ __forceinline__ v16bf load_a_bf(const bf16_t* __restrict__ p, int half) {
    v8bf lo = *(const v8bf*)(p + 8 * half);
    v8bf hi = *(const v8bf*)(p + 16 + 8 * half);
    return __builtin_shufflevector(lo, hi, 0, 1, 2, 3, 4, 5, 6, 7,
                                   8, 9, 10, 11, 12, 13, 14, 15);
}
// B-operand loader (16-bit B 32x16 layout): lane holds col = lane%16.
// Element e<8 -> K = 8*half + e ; e>=8 -> K = 16 + 8*half + (e-8).
__device__ __forceinline__ v16bf load_b_bf(const bf16_t* __restrict__ p, int half) {
    v8bf lo = *(const v8bf*)(p + 8 * half);
    v8bf hi = *(const v8bf*)(p + 16 + 8 * half);
    return __builtin_shufflevector(lo, hi, 0, 1, 2, 3, 4, 5, 6, 7,
                                   8, 9, 10, 11, 12, 13, 14, 15);
}

#define WMMA_BF16(A, B, C) \
    __builtin_amdgcn_wmma_f32_16x16x32_bf16(false, (A), false, (B), (short)0, (C), false, false)

// ---------------------------------------------------------------------------
// Elementwise fp32 -> bf16 cast, 8 elements/thread, 16B stores
// ---------------------------------------------------------------------------
__global__ __launch_bounds__(256) void cast_bf16(const float* __restrict__ s,
                                                 bf16_t* __restrict__ d, int n8) {
    const int i = blockIdx.x * blockDim.x + threadIdx.x;
    if (i >= n8) return;
    const size_t base = (size_t)i * 8;
    float4 f0 = *(const float4*)(s + base);
    float4 f1 = *(const float4*)(s + base + 4);
    uint4 o;
    o.x = (unsigned)f2bfu(f0.x) | ((unsigned)f2bfu(f0.y) << 16);
    o.y = (unsigned)f2bfu(f0.z) | ((unsigned)f2bfu(f0.w) << 16);
    o.z = (unsigned)f2bfu(f1.x) | ((unsigned)f2bfu(f1.y) << 16);
    o.w = (unsigned)f2bfu(f1.z) | ((unsigned)f2bfu(f1.w) << 16);
    *(uint4*)(d + base) = o;
}

// ---------------------------------------------------------------------------
// GEMM: C[M,N] = A[M,K] * W[N,K]^T, bf16 inputs, OutT output.
// 8 waves arranged 4(M) x 2(N); wave tile 32x64; block tile 128x128.
// ---------------------------------------------------------------------------
template <typename OutT>
__global__ __launch_bounds__(256) void gemm_bf16(const bf16_t* __restrict__ A,
                                                 const bf16_t* __restrict__ Bm,
                                                 OutT* __restrict__ C,
                                                 int M, int N, int K) {
    const int lane = threadIdx.x & 31;
    const int wv   = threadIdx.x >> 5;
    const int half = lane >> 4;
    const int r16  = lane & 15;
    const int m0 = blockIdx.y * 128 + (wv >> 1) * 32;
    const int n0 = blockIdx.x * 128 + (wv & 1) * 64;

    v8f acc[2][4] = {};

    const bf16_t* ar0 = A + (size_t)(m0 + r16) * K;
    const bf16_t* ar1 = A + (size_t)(m0 + 16 + r16) * K;
    const bf16_t* br[4];
#pragma unroll
    for (int j = 0; j < 4; ++j) br[j] = Bm + (size_t)(n0 + j * 16 + r16) * K;

    for (int k = 0; k < K; k += 32) {
        __builtin_prefetch(ar0 + k + 256, 0, 1);
        __builtin_prefetch(br[0] + k + 256, 0, 1);
        v16bf a0 = load_a_bf(ar0 + k, half);
        v16bf a1 = load_a_bf(ar1 + k, half);
#pragma unroll
        for (int j = 0; j < 4; ++j) {
            v16bf b = load_b_bf(br[j] + k, half);
            acc[0][j] = WMMA_BF16(a0, b, acc[0][j]);
            acc[1][j] = WMMA_BF16(a1, b, acc[1][j]);
        }
    }

    // C/D layout: VGPR v, lane l -> row v + 8*(l/16), col l%16
#pragma unroll
    for (int i = 0; i < 2; ++i)
#pragma unroll
        for (int j = 0; j < 4; ++j) {
            const int rowb = m0 + i * 16 + half * 8;
            const int col  = n0 + j * 16 + r16;
#pragma unroll
            for (int v = 0; v < 8; ++v) {
                const float val = acc[i][j][v];
                if constexpr (__is_same(OutT, float))
                    C[(size_t)(rowb + v) * N + col] = val;
                else
                    C[(size_t)(rowb + v) * N + col] = f2bf(val);
            }
        }
}

// ---------------------------------------------------------------------------
// RoPE applied in-place to bf16 Q and K ([M, D] layout, head-major rows).
// ---------------------------------------------------------------------------
__global__ __launch_bounds__(256) void rope_qk(bf16_t* __restrict__ Q,
                                               bf16_t* __restrict__ Kb) {
    const int idx = blockIdx.x * blockDim.x + threadIdx.x;
    const int i = idx & 63;
    const int h = (idx >> 6) & (CH - 1);
    const int m = idx >> 10;
    if (m >= CM) return;
    const int s = m & (CS - 1);

    const float inv = __expf(-9.2103403719761836f * ((float)i * (1.0f / 64.0f)));
    const float f = (float)s * inv;
    const float c = __cosf(f);
    const float sn = __sinf(f);

    const size_t base = (size_t)m * CD + h * CDH + i;
    float q1 = bf2f(Q[base]), q2 = bf2f(Q[base + 64]);
    Q[base]      = f2bf(q1 * c - q2 * sn);
    Q[base + 64] = f2bf(q2 * c + q1 * sn);
    float k1 = bf2f(Kb[base]), k2 = bf2f(Kb[base + 64]);
    Kb[base]      = f2bf(k1 * c - k2 * sn);
    Kb[base + 64] = f2bf(k2 * c + k1 * sn);
}

// ---------------------------------------------------------------------------
// V transpose: Vh bf16 [M, D] -> Vt bf16 [B*H, DH, S]  (LDS-tiled 32x32)
// grid = (S/32, DH/32, B*H), block = 256 (32x8)
// ---------------------------------------------------------------------------
__global__ __launch_bounds__(256) void transpose_v(const bf16_t* __restrict__ Vh,
                                                   bf16_t* __restrict__ Vt) {
    __shared__ u16 tile[32 * 33];
    const int tx = threadIdx.x & 31;
    const int ty = threadIdx.x >> 5; // 0..7
    const int bh = blockIdx.z;
    const int b  = bh >> 4;
    const int h  = bh & (CH - 1);
    const int s0  = blockIdx.x * 32;
    const int dh0 = blockIdx.y * 32;

#pragma unroll
    for (int i = 0; i < 4; ++i) {
        const int sl = ty + 8 * i;
        tile[sl * 33 + tx] = __builtin_bit_cast(
            u16, Vh[((size_t)(b * CS + s0 + sl)) * CD + h * CDH + dh0 + tx]);
    }
    __syncthreads();
#pragma unroll
    for (int i = 0; i < 4; ++i) {
        const int dl = ty + 8 * i;
        Vt[((size_t)bh * CDH + dh0 + dl) * CS + s0 + tx] =
            __builtin_bit_cast(bf16_t, tile[tx * 33 + dl]);
    }
}

// ---------------------------------------------------------------------------
// Flash attention: grid = (S/128, B*H). 8 waves/block, each wave owns 16 query
// rows x DH=128, streams K/V in 32-column tiles with online softmax. All bf16.
// ---------------------------------------------------------------------------
__global__ __launch_bounds__(256) void flash_attn(const bf16_t* __restrict__ Q,
                                                  const bf16_t* __restrict__ Km,
                                                  const bf16_t* __restrict__ Vt,
                                                  const float* __restrict__ bias,
                                                  bf16_t* __restrict__ O) {
    __shared__ u16 pbuf[8 * 16 * 34]; // per-wave 16x32 bf16 P tile, stride 34

    const int lane = threadIdx.x & 31;
    const int wv   = threadIdx.x >> 5;
    const int half = lane >> 4;
    const int r16  = lane & 15;

    const int bh = blockIdx.y;
    const int b  = bh >> 4;
    const int h  = bh & (CH - 1);
    const int q0 = blockIdx.x * 128 + wv * 16;
    const size_t mbase = (size_t)b * CS;

    const float mb    = bias[h];
    const float scale = 0.08838834764831845f; // 1/sqrt(128)

    const bf16_t* qrow = Q + (mbase + q0 + r16) * CD + h * CDH;
    v16bf aq[4];
#pragma unroll
    for (int c = 0; c < 4; ++c) aq[c] = load_a_bf(qrow + 32 * c, half);

    v8f o[8] = {};
    float mrow[8], lrow[8]; // per-row running max / sum (row = v + 8*half)
#pragma unroll
    for (int v = 0; v < 8; ++v) { mrow[v] = -3.0e38f; lrow[v] = 0.0f; }

    u16* myp = pbuf + wv * (16 * 34);
    const bf16_t* kbase0 = Km + (mbase + r16) * CD + h * CDH;
    const bf16_t* kbase1 = Km + (mbase + 16 + r16) * CD + h * CDH;

    for (int k0 = 0; k0 < CS; k0 += 32) {
        // ---- scores: two 16x16 tiles over DH=128 (4 bf16 WMMAs each) ----
        v8f s0 = {}, s1 = {};
        const bf16_t* kr0 = kbase0 + (size_t)k0 * CD;
        const bf16_t* kr1 = kbase1 + (size_t)k0 * CD;
#pragma unroll
        for (int c = 0; c < 4; ++c) {
            v16bf bk0 = load_b_bf(kr0 + 32 * c, half);
            v16bf bk1 = load_b_bf(kr1 + 32 * c, half);
            s0 = WMMA_BF16(aq[c], bk0, s0);
            s1 = WMMA_BF16(aq[c], bk1, s1);
        }

        // ---- online softmax (per row; butterfly over 16-lane half-wave) ----
        float p0v[8], p1v[8];
#pragma unroll
        for (int v = 0; v < 8; ++v) {
            float x0 = s0[v] * scale + mb;
            float x1 = s1[v] * scale + mb;
            float t = fmaxf(x0, x1);
            t = fmaxf(t, __shfl_xor(t, 1, 32));
            t = fmaxf(t, __shfl_xor(t, 2, 32));
            t = fmaxf(t, __shfl_xor(t, 4, 32));
            t = fmaxf(t, __shfl_xor(t, 8, 32));
            const float mnew = fmaxf(mrow[v], t);
            const float p0 = __expf(x0 - mnew);
            const float p1 = __expf(x1 - mnew);
            float ps = p0 + p1;
            ps += __shfl_xor(ps, 1, 32);
            ps += __shfl_xor(ps, 2, 32);
            ps += __shfl_xor(ps, 4, 32);
            ps += __shfl_xor(ps, 8, 32);
            const float corr = __expf(mrow[v] - mnew);
            lrow[v] = lrow[v] * corr + ps;
            mrow[v] = mnew;
            p0v[v] = p0;
            p1v[v] = p1;
#pragma unroll
            for (int t2 = 0; t2 < 8; ++t2) o[t2][v] *= corr;
        }

        // ---- P: C layout -> bf16 A layout via per-wave LDS round trip ----
#pragma unroll
        for (int v = 0; v < 8; ++v) {
            myp[(v + 8 * half) * 34 + r16]      = f2bfu(p0v[v]);
            myp[(v + 8 * half) * 34 + 16 + r16] = f2bfu(p1v[v]);
        }
        asm volatile("s_wait_dscnt 0" ::: "memory");
        // Element pairs (2j, 2j+1) sit at adjacent even columns -> read b32
        struct U8 { unsigned w[8]; } pw;
#pragma unroll
        for (int j = 0; j < 8; ++j) {
            const int col0 = (j < 4) ? (2 * j + 8 * half) : (16 + 2 * (j - 4) + 8 * half);
            pw.w[j] = *(const unsigned*)(myp + r16 * 34 + col0);
        }
        v16bf pa = __builtin_bit_cast(v16bf, pw);

        // ---- O += P(16x32) x V(32x128): 8 WMMAs, contiguous Vt rows ----
#pragma unroll
        for (int t2 = 0; t2 < 8; ++t2) {
            const bf16_t* vtr = Vt + ((size_t)bh * CDH + t2 * 16 + r16) * CS + k0;
            v16bf bv = load_b_bf(vtr, half);
            o[t2] = WMMA_BF16(pa, bv, o[t2]);
        }
    }

    // ---- normalize and write attention output (bf16 [M, D]) ----
#pragma unroll
    for (int t2 = 0; t2 < 8; ++t2) {
#pragma unroll
        for (int v = 0; v < 8; ++v) {
            const float val = o[t2][v] / lrow[v];
            O[(mbase + q0 + v + 8 * half) * CD + h * CDH + t2 * 16 + r16] = f2bf(val);
        }
    }
}

// ---------------------------------------------------------------------------
// Host-side launch
// ---------------------------------------------------------------------------
extern "C" void kernel_launch(void* const* d_in, const int* in_sizes, int n_in,
                              void* d_out, int out_size, void* d_ws, size_t ws_size,
                              hipStream_t stream) {
    (void)in_sizes; (void)n_in; (void)out_size; (void)ws_size;
    const float* X     = (const float*)d_in[0];
    const float* Wq    = (const float*)d_in[1];
    const float* Wk    = (const float*)d_in[2];
    const float* Wv    = (const float*)d_in[3];
    const float* Wo    = (const float*)d_in[4];
    const float* mbias = (const float*)d_in[5];
    float* out = (float*)d_out;

    const size_t MD = (size_t)CM * CD; // 8.4M
    const size_t DD = (size_t)CD * CD; // 4.2M
    bf16_t* Xh  = (bf16_t*)d_ws;
    bf16_t* Wqh = Xh + MD;
    bf16_t* Wkh = Wqh + DD;
    bf16_t* Wvh = Wkh + DD;
    bf16_t* Woh = Wvh + DD;
    bf16_t* Qh  = Woh + DD;
    bf16_t* Kh  = Qh + MD;
    bf16_t* Vh  = Kh + MD;
    bf16_t* Vtb = Vh + MD;
    bf16_t* Ah  = Vtb + MD;

    // 1) one-time bf16 casts
    cast_bf16<<<(int)(MD / 8 / 256), 256, 0, stream>>>(X, Xh, (int)(MD / 8));
    cast_bf16<<<(int)(DD / 8 / 256), 256, 0, stream>>>(Wq, Wqh, (int)(DD / 8));
    cast_bf16<<<(int)(DD / 8 / 256), 256, 0, stream>>>(Wk, Wkh, (int)(DD / 8));
    cast_bf16<<<(int)(DD / 8 / 256), 256, 0, stream>>>(Wv, Wvh, (int)(DD / 8));
    cast_bf16<<<(int)(DD / 8 / 256), 256, 0, stream>>>(Wo, Woh, (int)(DD / 8));

    // 2) projections (bf16 in, bf16 out)
    dim3 gg(CD / 128, CM / 128); // (16, 32)
    gemm_bf16<bf16_t><<<gg, 256, 0, stream>>>(Xh, Wqh, Qh, CM, CD, CD);
    gemm_bf16<bf16_t><<<gg, 256, 0, stream>>>(Xh, Wkh, Kh, CM, CD, CD);
    gemm_bf16<bf16_t><<<gg, 256, 0, stream>>>(Xh, Wvh, Vh, CM, CD, CD);

    // 3) RoPE in-place on Q/K
    rope_qk<<<(CM * CH * 64) / 256, 256, 0, stream>>>(Qh, Kh);

    // 4) V transpose for contiguous P*V B-operand loads
    transpose_v<<<dim3(CS / 32, CDH / 32, CB * CH), 256, 0, stream>>>(Vh, Vtb);

    // 5) flash attention
    flash_attn<<<dim3(CS / 128, CB * CH), 256, 0, stream>>>(Qh, Kh, Vtb, mbias, Ah);

    // 6) output projection (bf16 in, f32 out)
    gemm_bf16<float><<<gg, 256, 0, stream>>>(Ah, Woh, out, CM, CD, CD);
}
